// MultiHeadHierarchicalAttention_31645319037583
// MI455X (gfx1250) — matrix operand order repaired and checked by reading, not compiled, not run on hardware
//
#include <hip/hip_runtime.h>

// ---------------------------------------------------------------------------
// Problem constants (match reference)
// ---------------------------------------------------------------------------
static constexpr int B_  = 4;
static constexpr int LQ  = 256;
static constexpr int NS  = 24;
static constexpr int NT  = 128;
static constexpr int D   = 512;
static constexpr int H   = 8;
static constexpr int DK  = 64;

typedef __attribute__((ext_vector_type(16))) __bf16 v16bf;
typedef __attribute__((ext_vector_type(8)))  __bf16 bf16x8;
typedef __attribute__((ext_vector_type(4)))  __bf16 bf16x4;
typedef __attribute__((ext_vector_type(2)))  __bf16 bf16x2;
typedef __attribute__((ext_vector_type(8)))  float  v8f;

static __device__ inline bf16x4 cvt4(float4 v)
{
    bf16x4 r;
    r[0] = (__bf16)v.x; r[1] = (__bf16)v.y;
    r[2] = (__bf16)v.z; r[3] = (__bf16)v.w;
    return r;
}
static __device__ inline v16bf cat8(bf16x8 lo, bf16x8 hi)
{
    return __builtin_shufflevector(lo, hi, 0, 1, 2, 3, 4, 5, 6, 7,
                                   8, 9, 10, 11, 12, 13, 14, 15);
}
static __device__ inline v8f wmma_bf16(v16bf a, v16bf b, v8f c)
{
    return __builtin_amdgcn_wmma_f32_16x16x32_bf16(
               false, a, false, b, (short)0, c, false, false);
}
static __device__ inline void stc(float* p, float v)  { *p = v; }
static __device__ inline void stc(__bf16* p, float v) { *p = (__bf16)v; }

// CDNA5 async global->LDS 16B copy (ASYNCcnt-tracked; §15.18 opcode 98).
// lds must point into __shared__; low 32 bits of the flat address are the
// LDS byte offset.
static __device__ inline void async_copy_b128(void* lds, const void* gaddr)
{
    unsigned lds_off = (unsigned)(size_t)lds;
    asm volatile("global_load_async_to_lds_b128 %0, %1, off"
                 :: "v"(lds_off), "v"(gaddr)
                 : "memory");
}
static __device__ inline void async_wait0()
{
    asm volatile("s_wait_asynccnt 0x0" ::: "memory");
}

// ---------------------------------------------------------------------------
// Flat f32 -> bf16 conversion (n % 4 == 0)
// ---------------------------------------------------------------------------
__global__ void cvt_f32_bf16(const float* __restrict__ src,
                             __bf16* __restrict__ dst, int n4)
{
    int i = blockIdx.x * blockDim.x + threadIdx.x;
    if (i < n4)
        *(bf16x4*)&dst[i * 4] = cvt4(*(const float4*)&src[i * 4]);
}

// ---------------------------------------------------------------------------
// Weight transpose + convert:  Wt[n][k] = (bf16) W[k][n].   K,N % 32 == 0.
// One 32x32 tile per block (256 threads).
// ---------------------------------------------------------------------------
__global__ __launch_bounds__(256)
void transpose_cvt(const float* __restrict__ W, __bf16* __restrict__ Wt,
                   int K, int N)
{
    __shared__ float tile[32][33];
    const int kb = blockIdx.y * 32, nb = blockIdx.x * 32;
    const int tid = threadIdx.x;
    {
        int r = tid >> 3, c4 = tid & 7;            // 32 rows x 8 float4
        *(float4*)&tile[r][c4 * 4] =
            *(const float4*)&W[(size_t)(kb + r) * N + nb + c4 * 4];
    }
    __syncthreads();
    {
        int r = tid >> 3, c4 = tid & 7;            // 32 out-rows x 8 quads
        bf16x4 o;
        o[0] = (__bf16)tile[c4 * 4 + 0][r];
        o[1] = (__bf16)tile[c4 * 4 + 1][r];
        o[2] = (__bf16)tile[c4 * 4 + 2][r];
        o[3] = (__bf16)tile[c4 * 4 + 3][r];
        *(bf16x4*)&Wt[(size_t)(nb + r) * K + kb + c4 * 4] = o;
    }
}

// ---------------------------------------------------------------------------
// GEMM:  C[M,N] = A[M,K](bf16) @ Wt[N,K](bf16, pre-transposed) + bias[N]
// No LDS, no barriers: every WMMA fragment is two contiguous 16B runs in
// global bf16; operand reuse comes from WGP$/L2 (weights <=1MB, resident in
// the 192MB L2). 256 threads = 8 waves as 2x4; each wave computes 2x2 16x16
// tiles (32x32). Block tile 64x128. K%64==0, N%128==0; M guarded.
// k-loop unrolled x2 -> 8 WMMAs / 16 b128 loads per iteration.
// ---------------------------------------------------------------------------
template <typename OutT>
__global__ __launch_bounds__(256)
void gemm_bias_bf16(const __bf16* __restrict__ A,
                    const __bf16* __restrict__ Wt,
                    const float* __restrict__ bias,
                    OutT* __restrict__ C, int ldc,
                    int M, int N, int K)
{
    const int tid   = threadIdx.x;
    const int wave  = tid >> 5;
    const int lane  = tid & 31;
    const int half  = lane >> 4;
    const int l16   = lane & 15;
    const int mwave = wave >> 2;                 // 0..1
    const int nwave = wave & 3;                  // 0..3
    const int m0    = blockIdx.y * 64 + mwave * 32;
    const int n0    = blockIdx.x * 128 + nwave * 32;

    int rA0 = m0 + l16;      if (rA0 > M - 1) rA0 = M - 1;   // clamp (M=96 case)
    int rA1 = m0 + 16 + l16; if (rA1 > M - 1) rA1 = M - 1;
    const __bf16* a0 = A + (size_t)rA0 * K;
    const __bf16* a1 = A + (size_t)rA1 * K;
    const __bf16* b0 = Wt + (size_t)(n0 + l16) * K;
    const __bf16* b1 = Wt + (size_t)(n0 + 16 + l16) * K;

    v8f acc00 = {}, acc01 = {}, acc10 = {}, acc11 = {};

    for (int k0 = 0; k0 < K; k0 += 64) {
        #pragma unroll
        for (int u = 0; u < 2; ++u) {
            const int k = k0 + u * 32;
            v16bf af0 = cat8(*(const bf16x8*)(a0 + k + half * 8),
                             *(const bf16x8*)(a0 + k + 16 + half * 8));
            v16bf af1 = cat8(*(const bf16x8*)(a1 + k + half * 8),
                             *(const bf16x8*)(a1 + k + 16 + half * 8));
            v16bf bf0 = cat8(*(const bf16x8*)(b0 + k + half * 8),
                             *(const bf16x8*)(b0 + k + 16 + half * 8));
            v16bf bf1 = cat8(*(const bf16x8*)(b1 + k + half * 8),
                             *(const bf16x8*)(b1 + k + 16 + half * 8));
            acc00 = wmma_bf16(af0, bf0, acc00);
            acc01 = wmma_bf16(af0, bf1, acc01);
            acc10 = wmma_bf16(af1, bf0, acc10);
            acc11 = wmma_bf16(af1, bf1, acc11);
        }
    }

    // epilogue: C layout row = r + 8*half, col = l16 within each 16x16 tile
    #pragma unroll
    for (int mi = 0; mi < 2; ++mi) {
        v8f accn0 = mi ? acc10 : acc00;
        v8f accn1 = mi ? acc11 : acc01;
        const int row0 = m0 + mi * 16 + half * 8;
        #pragma unroll
        for (int ni = 0; ni < 2; ++ni) {
            v8f a = ni ? accn1 : accn0;
            const int col = n0 + ni * 16 + l16;
            const float bv = bias[col];
            #pragma unroll
            for (int r = 0; r < 8; ++r) {
                int row = row0 + r;
                if (row < M)
                    stc(C + (size_t)row * ldc + col, a[r] + bv);
            }
        }
    }
}

// ---------------------------------------------------------------------------
// Sentence-level graph attention. One block per (b,h); one thread per query.
// bf16 inputs, f32 math. Produces attns[B,H,LQ,NS] (f32) and ctx_s (bf16,
// laid out [B,LQ,H*DK]).
// ---------------------------------------------------------------------------
__global__ __launch_bounds__(256)
void sent_attn(const __bf16* __restrict__ qs, const __bf16* __restrict__ ks,
               const __bf16* __restrict__ vs, const float* __restrict__ bias_s,
               const float* __restrict__ graph,
               float* __restrict__ attns_out, __bf16* __restrict__ ctx_s)
{
    const int b = blockIdx.x / H;
    const int h = blockIdx.x % H;

    __shared__ float Kls[NS][DK];
    __shared__ float Vls[NS][DK];
    __shared__ float Gls[NS][NS];

    for (int i = threadIdx.x; i < NS * DK / 4; i += 256) {
        int s = i / (DK / 4), d4 = i % (DK / 4);
        size_t off = ((size_t)b * NS + s) * D + h * DK + d4 * 4;
        bf16x4 kv = *(const bf16x4*)&ks[off];
        bf16x4 vv = *(const bf16x4*)&vs[off];
        #pragma unroll
        for (int j = 0; j < 4; ++j) {
            Kls[s][d4 * 4 + j] = (float)kv[j];
            Vls[s][d4 * 4 + j] = (float)vv[j];
        }
    }
    for (int i = threadIdx.x; i < NS * NS; i += 256)
        Gls[i / NS][i % NS] = graph[(((size_t)b * H + h) * NS + i / NS) * NS + (i % NS)];
    __syncthreads();

    const int q = threadIdx.x;           // LQ == 256 == blockDim.x
    float qr[DK];
    const __bf16* qp = qs + ((size_t)b * LQ + q) * D + h * DK;
    #pragma unroll
    for (int d8 = 0; d8 < DK / 8; ++d8) {
        bf16x8 v = *(const bf16x8*)(qp + d8 * 8);
        #pragma unroll
        for (int j = 0; j < 8; ++j) qr[d8 * 8 + j] = (float)v[j];
    }

    const float scale = 0.125f;          // DK^-0.5
    const float* bsp = bias_s + (((size_t)b * H + h) * LQ + q) * NS;
    float a[NS], p[NS], a2[NS];
    float mx = -1e30f;
    for (int s = 0; s < NS; ++s) {
        float dot = 0.f;
        #pragma unroll 8
        for (int d = 0; d < DK; ++d) dot += qr[d] * Kls[s][d];
        a[s] = dot * scale + bsp[s];
        mx = fmaxf(mx, a[s]);
    }
    float sum = 0.f;
    for (int s = 0; s < NS; ++s) { p[s] = __expf(a[s] - mx); sum += p[s]; }
    const float inv = 1.0f / sum;

    // gaussian graph prior: (0.5*POS_WIN)^2 == 1.0
    float mx2 = -1e30f;
    for (int t = 0; t < NS; ++t) {
        float g = 0.f;
        for (int s = 0; s < NS; ++s) g += p[s] * inv * Gls[s][t];
        float val = a[t] - 0.5f * g * g;
        a2[t] = val;
        mx2 = fmaxf(mx2, val);
    }
    float sum2 = 0.f;
    for (int s = 0; s < NS; ++s) { a2[s] = __expf(a2[s] - mx2); sum2 += a2[s]; }
    const float inv2 = 1.0f / sum2;
    float* ap = attns_out + (((size_t)b * H + h) * LQ + q) * NS;
    for (int s = 0; s < NS; ++s) {
        a2[s] *= inv2;
        ap[s] = a2[s];
    }
    __bf16* cp = ctx_s + ((size_t)b * LQ + q) * D + h * DK;
    for (int d = 0; d < DK; ++d) {
        float c = 0.f;
        for (int s = 0; s < NS; ++s) c += a2[s] * Vls[s][d];
        cp[d] = (__bf16)c;
    }
}

// ---------------------------------------------------------------------------
// Word-level attention with WMMA. One block per (b,s,h); 8 waves, each wave
// owns 32 query rows (2 M-tiles). All operands already bf16 in global.
// K-tile staging uses CDNA5 GLOBAL_LOAD_ASYNC_TO_LDS_B128 (ASYNCcnt);
// softmax via shfl_xor butterfly over the 16-lane C-matrix row groups.
//   Kls[t][d]     : QK^T B-frag runs d = c*32 + (0|16) + half*8 + {0..7}
//   Vt[d][t]      : PV   B-frag runs t = c*32 + (0|16) + half*8 + {0..7}
//   Pbuf[w][r][t] : PV   A-frag runs t = c*32 + (0|16) + half*8 + {0..7}
//   Q A-frags load directly from global bf16 (two b128 per chunk).
// 64 KB static LDS.
// ---------------------------------------------------------------------------
__global__ __launch_bounds__(256)
void word_attn(const __bf16* __restrict__ qw, const __bf16* __restrict__ kw,
               const __bf16* __restrict__ vw, const float* __restrict__ bias_w,
               const float* __restrict__ attns, float* __restrict__ ctx_w)
{
    const int h = blockIdx.x % H;
    const int s = (blockIdx.x / H) % NS;
    const int b = blockIdx.x / (H * NS);

    __shared__ __bf16 Kls[NT][DK];          // 16 KB
    __shared__ __bf16 Vt[DK][NT];           // 16 KB (transposed)
    __shared__ __bf16 Pbuf[8][16][NT];      // 32 KB (per-wave P tile)

    const int tid  = threadIdx.x;
    const int wave = tid >> 5;
    const int lane = tid & 31;
    const int half = lane >> 4;
    const int l16  = lane & 15;

    const __bf16* kbase = kw + (size_t)(b * NS + s) * NT * D + h * DK;
    const __bf16* vbase = vw + (size_t)(b * NS + s) * NT * D + h * DK;

    // K: async 16B global->LDS copies (no VGPR round-trip), hidden behind
    // the V transpose staging below.
    #pragma unroll
    for (int i = 0; i < 4; ++i) {
        int idx = tid + i * 256;            // 0..1023
        int t = idx >> 3, d8 = idx & 7;
        async_copy_b128(&Kls[t][d8 * 8], kbase + (size_t)t * D + d8 * 8);
    }
    // V: transposed; two key-rows packed -> b32 LDS stores
    #pragma unroll
    for (int i = 0; i < 2; ++i) {
        int idx = tid + i * 256;            // 0..511
        int t2 = idx >> 3, d8 = idx & 7;    // key pair, dim octet
        int t = t2 * 2;
        bf16x8 r0 = *(const bf16x8*)(vbase + (size_t)t * D + d8 * 8);
        bf16x8 r1 = *(const bf16x8*)(vbase + (size_t)(t + 1) * D + d8 * 8);
        #pragma unroll
        for (int j = 0; j < 8; ++j) {
            bf16x2 pr; pr[0] = r0[j]; pr[1] = r1[j];
            *(bf16x2*)&Vt[d8 * 8 + j][t] = pr;
        }
    }
    async_wait0();          // this wave's async K copies landed in LDS
    __syncthreads();        // all waves' staging visible

    const float scale = 0.125f;

    for (int mt = 0; mt < 2; ++mt) {
        const int qrow0 = wave * 32 + mt * 16;

        // ---- Q A-fragments: direct global bf16, two b128 per chunk ----
        const __bf16* qrow = qw + ((size_t)b * LQ + qrow0 + l16) * D + h * DK;
        v16bf qa[2];
        #pragma unroll
        for (int c = 0; c < 2; ++c)
            qa[c] = cat8(*(const bf16x8*)(qrow + c * 32 + half * 8),
                         *(const bf16x8*)(qrow + c * 32 + 16 + half * 8));

        // ---- S = scale * Q K^T + bias_w  (8 N-tiles of 16 keys) ----
        const float* bwp = bias_w +
            ((((size_t)b * NS + s) * H + h) * LQ + qrow0 + half * 8) * NT + l16;
        v8f st[8];
        #pragma unroll
        for (int nt = 0; nt < 8; ++nt) {
            v8f acc = {};
            const int key = nt * 16 + l16;
            #pragma unroll
            for (int c = 0; c < 2; ++c) {
                v16bf bfrag = cat8(*(const bf16x8*)&Kls[key][c * 32 + half * 8],
                                   *(const bf16x8*)&Kls[key][c * 32 + 16 + half * 8]);
                acc = wmma_bf16(qa[c], bfrag, acc);
            }
            #pragma unroll
            for (int r = 0; r < 8; ++r)
                st[nt][r] = acc[r] * scale + bwp[(size_t)r * NT + nt * 16];
        }

        // ---- row softmax via butterfly reductions over 16-lane groups ----
        float mxr[8], invr[8];
        #pragma unroll
        for (int r = 0; r < 8; ++r) {
            float m = -1e30f;
            #pragma unroll
            for (int nt = 0; nt < 8; ++nt) m = fmaxf(m, st[nt][r]);
            #pragma unroll
            for (int dlt = 1; dlt < 16; dlt <<= 1)
                m = fmaxf(m, __shfl_xor(m, dlt, 32));
            float sm = 0.f;
            #pragma unroll
            for (int nt = 0; nt < 8; ++nt) sm += __expf(st[nt][r] - m);
            #pragma unroll
            for (int dlt = 1; dlt < 16; dlt <<= 1)
                sm += __shfl_xor(sm, dlt, 32);
            mxr[r] = m;
            invr[r] = 1.0f / sm;
        }
        // store normalized P (bf16) to per-wave LDS tile
        #pragma unroll
        for (int nt = 0; nt < 8; ++nt)
            #pragma unroll
            for (int r = 0; r < 8; ++r)
                Pbuf[wave][r + 8 * half][nt * 16 + l16] =
                    (__bf16)(__expf(st[nt][r] - mxr[r]) * invr[r]);

        // ---- O = P V  (A from Pbuf, B from Vt), scaled by sentence attn ----
        v16bf pa[4];
        #pragma unroll
        for (int c = 0; c < 4; ++c)
            pa[c] = cat8(*(const bf16x8*)&Pbuf[wave][l16][c * 32 + half * 8],
                         *(const bf16x8*)&Pbuf[wave][l16][c * 32 + 16 + half * 8]);

        float wsent[8];
        {
            const float* ap = attns +
                (((size_t)b * H + h) * LQ + qrow0 + half * 8) * NS + s;
            #pragma unroll
            for (int r = 0; r < 8; ++r)
                wsent[r] = ap[(size_t)r * NS];
        }
        #pragma unroll
        for (int nt = 0; nt < 4; ++nt) {
            v8f acc = {};
            const int dcol = nt * 16 + l16;
            #pragma unroll
            for (int c = 0; c < 4; ++c) {
                v16bf bfrag = cat8(*(const bf16x8*)&Vt[dcol][c * 32 + half * 8],
                                   *(const bf16x8*)&Vt[dcol][c * 32 + 16 + half * 8]);
                acc = wmma_bf16(pa[c], bfrag, acc);
            }
            float* op = ctx_w + ((size_t)b * LQ + qrow0 + half * 8) * D + h * DK + dcol;
            #pragma unroll
            for (int r = 0; r < 8; ++r)
                atomicAdd(op + (size_t)r * D, acc[r] * wsent[r]);
        }
    }
}

__global__ void zero_f32(float* __restrict__ p, int n)
{
    int i = blockIdx.x * blockDim.x + threadIdx.x;
    if (i < n) p[i] = 0.0f;
}

// ---------------------------------------------------------------------------
// Launch pipeline
// ---------------------------------------------------------------------------
extern "C" void kernel_launch(void* const* d_in, const int* in_sizes, int n_in,
                              void* d_out, int out_size, void* d_ws, size_t ws_size,
                              hipStream_t stream)
{
    const float* q      = (const float*)d_in[0];
    const float* k_s    = (const float*)d_in[1];
    const float* v_s    = (const float*)d_in[2];
    const float* k_w    = (const float*)d_in[3];
    const float* v_w    = (const float*)d_in[4];
    const float* bias_w = (const float*)d_in[5];
    const float* bias_s = (const float*)d_in[6];
    const float* gbias  = (const float*)d_in[7];
    const float* wq_s   = (const float*)d_in[8];
    const float* wk_s   = (const float*)d_in[9];
    const float* wv_s   = (const float*)d_in[10];
    const float* wq_w   = (const float*)d_in[11];
    const float* wk_w   = (const float*)d_in[12];
    const float* wv_w   = (const float*)d_in[13];
    const float* fcs_w  = (const float*)d_in[14];
    const float* fcw_w  = (const float*)d_in[15];
    const float* bq_s   = (const float*)d_in[16];
    const float* bk_s   = (const float*)d_in[17];
    const float* bv_s   = (const float*)d_in[18];
    const float* bq_w   = (const float*)d_in[19];
    const float* bk_w   = (const float*)d_in[20];
    const float* bv_w   = (const float*)d_in[21];
    const float* fcs_b  = (const float*)d_in[22];
    const float* fcw_b  = (const float*)d_in[23];
    const float* fco_w  = (const float*)d_in[24];
    const float* fco_b  = (const float*)d_in[25];

    // ---- workspace carve-out (all chunk sizes are multiples of 16B) ----
    const size_t BLQD   = (size_t)B_ * LQ * D;         // 524288
    const size_t BNSD   = (size_t)B_ * NS * D;         // 49152
    const size_t BNSWD  = (size_t)B_ * NS * NT * D;    // 6291456
    const size_t BHLQNS = (size_t)B_ * H * LQ * NS;    // 196608
    const size_t WDD    = (size_t)D * D;               // 262144

    char* wp = (char*)d_ws;
    auto abf = [&](size_t n) { __bf16* p = (__bf16*)wp; wp += n * 2; return p; };
    auto af32 = [&](size_t n) { float* p = (float*)wp; wp += n * 4; return p; };

    __bf16* q_bf  = abf(BLQD);
    __bf16* ks_bf = abf(BNSD);
    __bf16* vs_bf = abf(BNSD);
    __bf16* kw_bf = abf(BNSWD);
    __bf16* vw_bf = abf(BNSWD);
    __bf16* wt_qs = abf(WDD);
    __bf16* wt_ks = abf(WDD);
    __bf16* wt_vs = abf(WDD);
    __bf16* wt_qw = abf(WDD);
    __bf16* wt_kw = abf(WDD);
    __bf16* wt_vw = abf(WDD);
    __bf16* wt_fcs = abf(WDD);
    __bf16* wt_fcw = abf(WDD);
    __bf16* wt_fco = abf(2 * WDD);
    __bf16* qs_p  = abf(BLQD);
    __bf16* qw_p  = abf(BLQD);
    __bf16* ksp   = abf(BNSD);
    __bf16* vsp   = abf(BNSD);
    __bf16* kwp   = abf(BNSWD);
    __bf16* vwp   = abf(BNSWD);
    __bf16* ctxs  = abf(BLQD);
    __bf16* ctxw_bf = abf(BLQD);
    __bf16* cat_p = abf(2 * BLQD);
    float*  att_p = af32(BHLQNS);
    float*  ctxw_f = af32(BLQD);

    const dim3 blk(256);
    auto cvt = [&](const float* s, __bf16* d, size_t n) {
        int n4 = (int)(n / 4);
        cvt_f32_bf16<<<(n4 + 255) / 256, blk, 0, stream>>>(s, d, n4);
    };
    auto tr = [&](const float* w, __bf16* wt, int K, int N) {
        transpose_cvt<<<dim3(N / 32, K / 32), blk, 0, stream>>>(w, wt, K, N);
    };
    auto ggrid = [](int M, int N) { return dim3(N / 128, (M + 63) / 64); };

    // ---- one-time conversions (half the stream bytes for everything after) ----
    cvt(q,   q_bf,  BLQD);
    cvt(k_s, ks_bf, BNSD);
    cvt(v_s, vs_bf, BNSD);
    cvt(k_w, kw_bf, BNSWD);
    cvt(v_w, vw_bf, BNSWD);
    tr(wq_s, wt_qs, D, D);  tr(wk_s, wt_ks, D, D);  tr(wv_s, wt_vs, D, D);
    tr(wq_w, wt_qw, D, D);  tr(wk_w, wt_kw, D, D);  tr(wv_w, wt_vw, D, D);
    tr(fcs_w, wt_fcs, D, D); tr(fcw_w, wt_fcw, D, D);
    tr(fco_w, wt_fco, 2 * D, D);

    // clear word-context accumulator (re-zeroed every call: deterministic)
    zero_f32<<<((int)BLQD + 255) / 256, blk, 0, stream>>>(ctxw_f, (int)BLQD);

    // ---- projections (LDS-free bf16 WMMA GEMMs) ----
    gemm_bias_bf16<<<ggrid(B_ * LQ, D), blk, 0, stream>>>(q_bf,  wt_qs, bq_s, qs_p, D, B_ * LQ, D, D);
    gemm_bias_bf16<<<ggrid(B_ * LQ, D), blk, 0, stream>>>(q_bf,  wt_qw, bq_w, qw_p, D, B_ * LQ, D, D);
    gemm_bias_bf16<<<ggrid(B_ * NS, D), blk, 0, stream>>>(ks_bf, wt_ks, bk_s, ksp,  D, B_ * NS, D, D);
    gemm_bias_bf16<<<ggrid(B_ * NS, D), blk, 0, stream>>>(vs_bf, wt_vs, bv_s, vsp,  D, B_ * NS, D, D);
    gemm_bias_bf16<<<ggrid(B_ * NS * NT, D), blk, 0, stream>>>(kw_bf, wt_kw, bk_w, kwp, D, B_ * NS * NT, D, D);
    gemm_bias_bf16<<<ggrid(B_ * NS * NT, D), blk, 0, stream>>>(vw_bf, wt_vw, bv_w, vwp, D, B_ * NS * NT, D, D);

    // ---- sentence graph attention -> attns + ctx_s ----
    sent_attn<<<B_ * H, blk, 0, stream>>>(qs_p, ksp, vsp, bias_s, gbias, att_p, ctxs);

    // ---- word attention (WMMA) -> ctx_w (atomic f32 accumulate over s) ----
    word_attn<<<B_ * NS * H, blk, 0, stream>>>(qw_p, kwp, vwp, bias_w, att_p, ctxw_f);
    cvt(ctxw_f, ctxw_bf, BLQD);

    // ---- fcs / fcw written side-by-side into concat buffer [B*LQ, 2D] ----
    gemm_bias_bf16<<<ggrid(B_ * LQ, D), blk, 0, stream>>>(ctxs,    wt_fcs, fcs_b, cat_p,     2 * D, B_ * LQ, D, D);
    gemm_bias_bf16<<<ggrid(B_ * LQ, D), blk, 0, stream>>>(ctxw_bf, wt_fcw, fcw_b, cat_p + D, 2 * D, B_ * LQ, D, D);

    // ---- final projection: [B*LQ, 2D] @ Wt + fco_b -> d_out (f32) ----
    gemm_bias_bf16<<<ggrid(B_ * LQ, D), blk, 0, stream>>>(cat_p, wt_fco, fco_b,
                                                          (float*)d_out, D, B_ * LQ, 2 * D, D);
}
